// PointNetEncoder_75076028334683
// MI455X (gfx1250) — matrix operand order repaired and checked by reading, not compiled, not run on hardware
//
#include <hip/hip_runtime.h>
#include <hip/hip_bf16.h>

// ---------------------------------------------------------------------------
// Types for CDNA5 WMMA (wave32): V_WMMA_F32_16X16X32_BF16
// ---------------------------------------------------------------------------
typedef __bf16 v16bf __attribute__((ext_vector_type(16)));
typedef __bf16 v8bf  __attribute__((ext_vector_type(8)));
typedef float  v8f   __attribute__((ext_vector_type(8)));
typedef float  v4f   __attribute__((ext_vector_type(4)));

#define BATCH 4
#define NPTS  8192            // points per batch
#define NPTOT (BATCH * NPTS)  // 32768
#define LDS_PAD 16            // bf16 elements of row padding (32B)

__device__ __forceinline__ unsigned short f2bfbits(float f) {
    unsigned u = __builtin_bit_cast(unsigned, f);
    unsigned r = u + 0x7FFFu + ((u >> 16) & 1u);   // RNE
    return (unsigned short)(r >> 16);
}
__device__ __forceinline__ __bf16 f2bf(float f) {
    unsigned short h = f2bfbits(f);
    return __builtin_bit_cast(__bf16, h);
}

__device__ __forceinline__ void atomicMaxFloat(float* addr, float val) {
    if (val >= 0.0f) {
        atomicMax((int*)addr, __float_as_int(val));
    } else {
        atomicMin((unsigned int*)addr, __float_as_uint(val));
    }
}

// ---------------------------------------------------------------------------
// fp32 -> bf16 weight conversion
// ---------------------------------------------------------------------------
__global__ void k_cvt_bf16(const float* __restrict__ src, __bf16* __restrict__ dst, int n) {
    int i = blockIdx.x * blockDim.x + threadIdx.x;
    if (i < n) dst[i] = f2bf(src[i]);
}

// ---------------------------------------------------------------------------
// Layer 1: 3 -> 64, ReLU. One thread per point; point-major bf16 [NP,64],
// packed 8 bf16 per 16B store.
// ---------------------------------------------------------------------------
__global__ void k_l1(const float* __restrict__ pc, const float* __restrict__ W1,
                     const float* __restrict__ b1, __bf16* __restrict__ hT) {
    int p = blockIdx.x * blockDim.x + threadIdx.x;     // 0..NPTOT-1
    if (p >= NPTOT) return;
    int b = p >> 13;                                   // /8192
    int m = p & (NPTS - 1);
    size_t base = (size_t)b * 3 * NPTS + m;
    float x = pc[base], y = pc[base + NPTS], z = pc[base + 2 * NPTS];
    uint4* row = (uint4*)(hT + (size_t)p * 64);
#pragma unroll
    for (int o8 = 0; o8 < 8; ++o8) {
        unsigned q[4];
#pragma unroll
        for (int j = 0; j < 4; ++j) {
            int o = o8 * 8 + 2 * j;
            float h0 = fmaf(W1[o * 3 + 0], x, fmaf(W1[o * 3 + 1], y, fmaf(W1[o * 3 + 2], z, b1[o])));
            float h1 = fmaf(W1[o * 3 + 3], x, fmaf(W1[o * 3 + 4], y, fmaf(W1[o * 3 + 5], z, b1[o + 1])));
            q[j] = (unsigned)f2bfbits(fmaxf(h0, 0.0f)) | ((unsigned)f2bfbits(fmaxf(h1, 0.0f)) << 16);
        }
        row[o8] = make_uint4(q[0], q[1], q[2], q[3]);
    }
}

// ---------------------------------------------------------------------------
// bf16 WMMA GEMM:  Y[oc, n] = act( W[oc, k] * Xt[n, k] + bias[oc] )
//   Block: stages one 16(n) x Cin activation tile into LDS via
//   global_load_async_to_lds_b128 (ASYNCcnt), then 4 waves x MT oc-tiles,
//   each reusing the LDS-resident B fragment across MT WMMAs.
//   relu!=0 : ReLU, store point-major bf16 Yt [Npts,Cout] (+ optional fp32)
//   relu==0 : linear, per-tile lane max-reduce, atomic fold into gmax[b,oc]
// ---------------------------------------------------------------------------
template <int MT>
__global__ void k_gemm_bf16(const __bf16* __restrict__ W, const float* __restrict__ bias,
                            const __bf16* __restrict__ Xt, int Cin, int Cout,
                            __bf16* __restrict__ Yt, float* __restrict__ Yf,
                            float* __restrict__ gmax, int relu, int Nper) {
    __shared__ alignas(64) __bf16 sX[16 * (256 + LDS_PAD)];
    const int stride = Cin + LDS_PAD;
    const int n0 = blockIdx.x * 16;

    // --- async stage: 16 rows x Cin bf16, 16B chunks -----------------------
    {
        const int cpr = Cin >> 3;          // 16B chunks per row
        const int nchunks = 16 * cpr;
        for (int c = threadIdx.x; c < nchunks; c += blockDim.x) {
            int row = c / cpr;
            int col = (c - row * cpr) * 8;
            unsigned long long ga =
                (unsigned long long)(Xt + (size_t)(n0 + row) * Cin + col);
            unsigned lds = (unsigned)(unsigned long long)(&sX[row * stride + col]);
            asm volatile("global_load_async_to_lds_b128 %0, %1, off"
                         :: "v"(lds), "v"(ga) : "memory");
        }
        asm volatile("s_wait_asynccnt 0x0" ::: "memory");
        __syncthreads();
    }

    const int lane = threadIdx.x & 31;
    const int wave = threadIdx.x >> 5;
    const int tile0 = (blockIdx.y * 4 + wave) * MT;   // first oc-tile of wave

    v8f acc[MT];
#pragma unroll
    for (int t = 0; t < MT; ++t) acc[t] = (v8f){};

    const int mA = lane & 15;
    const int ka = (lane < 16) ? 0 : 8;    // A: K chunks {ka..ka+7, ka+16..ka+23}
    const int kb = (lane < 16) ? 0 : 8;    // B: K chunks {kb..kb+7, kb+16..kb+23}
    const __bf16* xr = &sX[(lane & 15) * stride + kb];

    for (int k0 = 0; k0 < Cin; k0 += 32) {
        v8bf blo = *(const v8bf*)(xr + k0);            // ds_load from staged tile
        v8bf bhi = *(const v8bf*)(xr + k0 + 16);
        v16bf bm;
#pragma unroll
        for (int i = 0; i < 8; ++i) { bm[i] = blo[i]; bm[8 + i] = bhi[i]; }
#pragma unroll
        for (int t = 0; t < MT; ++t) {
            const __bf16* wr = W + (size_t)((tile0 + t) * 16 + mA) * Cin + k0 + ka;
            v8bf alo = *(const v8bf*)wr;
            v8bf ahi = *(const v8bf*)(wr + 16);
            v16bf a;
#pragma unroll
            for (int i = 0; i < 8; ++i) { a[i] = alo[i]; a[8 + i] = ahi[i]; }
            acc[t] = __builtin_amdgcn_wmma_f32_16x16x32_bf16(
                false, a, false, bm, (short)0, acc[t], false, false);
        }
    }

    const int half = (lane >> 4) << 3;     // 0 or 8 (M offset)
    const int n    = lane & 15;
    const int gn   = n0 + n;

    if (relu) {
#pragma unroll
        for (int t = 0; t < MT; ++t) {
#pragma unroll
            for (int r = 0; r < 8; ++r) {
                int oc = (tile0 + t) * 16 + r + half;
                float y = fmaxf(acc[t][r] + bias[oc], 0.0f);
                if (Yt) Yt[(size_t)gn * Cout + oc] = f2bf(y);
                if (Yf) Yf[(size_t)gn * Cout + oc] = y;
            }
        }
    } else {
#pragma unroll
        for (int t = 0; t < MT; ++t) {
#pragma unroll
            for (int r = 0; r < 8; ++r) {
                int oc = (tile0 + t) * 16 + r + half;
                float y = acc[t][r] + bias[oc];
                for (int off = 1; off < 16; off <<= 1)
                    y = fmaxf(y, __shfl_xor(y, off, 32));
                if (n == 0) atomicMaxFloat(&gmax[(size_t)(gn / Nper) * Cout + oc], y);
            }
        }
    }
}

// ---------------------------------------------------------------------------
// gmax init to -inf bit pattern
// ---------------------------------------------------------------------------
__global__ void k_init_gmax(unsigned* __restrict__ g, int n) {
    int i = blockIdx.x * blockDim.x + threadIdx.x;
    if (i < n) g[i] = 0xFF800000u;   // -inf
}

// ---------------------------------------------------------------------------
// Broadcast gmax into out channels [0,1024): out[b,c,m] = gmax[b,c]
// Non-temporal 16B stores: output is never re-read, keep L2 for activations.
// ---------------------------------------------------------------------------
__global__ void k_broadcast(const float* __restrict__ gmax, float* __restrict__ out) {
    const int NQ = NPTS / 4;
    long long idx = (long long)blockIdx.x * blockDim.x + threadIdx.x;
    long long total = (long long)BATCH * 1024 * NQ;
    if (idx >= total) return;
    int m4 = (int)(idx % NQ);
    int c  = (int)((idx / NQ) % 1024);
    int b  = (int)(idx / ((long long)1024 * NQ));
    float g = gmax[b * 1024 + c];
    v4f v = {g, g, g, g};
    __builtin_nontemporal_store(v, (v4f*)out + ((size_t)b * 1152 + c) * NQ + m4);
}

// ---------------------------------------------------------------------------
// Fused KNN (top-3 over squared distance, point cloud tiled through LDS)
// + gather/mean of local features into out channels [1024,1152)
// ---------------------------------------------------------------------------
__global__ void k_knn_gather(const float* __restrict__ pc, const float* __restrict__ qp,
                             const float* __restrict__ lF, float* __restrict__ out) {
    __shared__ float sx[2048], sy[2048], sz[2048];
    const int b = blockIdx.y;
    const int m = blockIdx.x * blockDim.x + threadIdx.x;
    const size_t base = (size_t)b * 3 * NPTS;
    const float qx = qp[base + m], qy = qp[base + NPTS + m], qz = qp[base + 2 * NPTS + m];

    float d0 = 3.4e38f, d1 = 3.4e38f, d2 = 3.4e38f;
    int   i0 = 0, i1 = 0, i2 = 0;

    for (int c0 = 0; c0 < NPTS; c0 += 2048) {
        __syncthreads();
        for (int i = threadIdx.x; i < 2048; i += blockDim.x) {
            sx[i] = pc[base + c0 + i];
            sy[i] = pc[base + NPTS + c0 + i];
            sz[i] = pc[base + 2 * NPTS + c0 + i];
        }
        __syncthreads();
#pragma unroll 4
        for (int j = 0; j < 2048; ++j) {
            float dx = qx - sx[j], dy = qy - sy[j], dz = qz - sz[j];
            float d = fmaf(dx, dx, fmaf(dy, dy, dz * dz));
            int idx = c0 + j;
            if (d < d2) {
                if (d < d1) {
                    if (d < d0) { d2 = d1; i2 = i1; d1 = d0; i1 = i0; d0 = d; i0 = idx; }
                    else        { d2 = d1; i2 = i1; d1 = d;  i1 = idx; }
                } else          { d2 = d;  i2 = idx; }
            }
        }
    }

    const float4* r0 = (const float4*)(lF + ((size_t)b * NPTS + i0) * 128);
    const float4* r1 = (const float4*)(lF + ((size_t)b * NPTS + i1) * 128);
    const float4* r2 = (const float4*)(lF + ((size_t)b * NPTS + i2) * 128);
    const float inv = 1.0f / 3.0f;
    size_t ob = (size_t)b * 1152 * NPTS + (size_t)1024 * NPTS + m;
#pragma unroll 8
    for (int c4 = 0; c4 < 32; ++c4) {
        float4 a = r0[c4], bb = r1[c4], cc = r2[c4];
        __builtin_nontemporal_store((a.x + bb.x + cc.x) * inv, &out[ob + (size_t)(4 * c4 + 0) * NPTS]);
        __builtin_nontemporal_store((a.y + bb.y + cc.y) * inv, &out[ob + (size_t)(4 * c4 + 1) * NPTS]);
        __builtin_nontemporal_store((a.z + bb.z + cc.z) * inv, &out[ob + (size_t)(4 * c4 + 2) * NPTS]);
        __builtin_nontemporal_store((a.w + bb.w + cc.w) * inv, &out[ob + (size_t)(4 * c4 + 3) * NPTS]);
    }
}

// ---------------------------------------------------------------------------
extern "C" void kernel_launch(void* const* d_in, const int* in_sizes, int n_in,
                              void* d_out, int out_size, void* d_ws, size_t ws_size,
                              hipStream_t stream) {
    (void)in_sizes; (void)n_in; (void)out_size; (void)ws_size;
    const float* pc = (const float*)d_in[0];
    const float* qp = (const float*)d_in[1];
    const float* W1 = (const float*)d_in[2];
    const float* b1 = (const float*)d_in[3];
    const float* W2 = (const float*)d_in[4];
    const float* b2 = (const float*)d_in[5];
    const float* G1 = (const float*)d_in[6];
    const float* g1 = (const float*)d_in[7];
    const float* G2 = (const float*)d_in[8];
    const float* g2 = (const float*)d_in[9];
    float* out = (float*)d_out;

    // workspace layout (256B aligned slices)
    char* ws = (char*)d_ws;
    size_t off = 0;
    auto nxt = [&](size_t bytes) { size_t r = off; off = (off + bytes + 255) & ~(size_t)255; return r; };
    __bf16* wW2 = (__bf16*)(ws + nxt(128 * 64 * 2));
    __bf16* wG1 = (__bf16*)(ws + nxt(256 * 128 * 2));
    __bf16* wG2 = (__bf16*)(ws + nxt(1024 * 256 * 2));
    __bf16* hT  = (__bf16*)(ws + nxt((size_t)NPTOT * 64 * 2));
    __bf16* lT  = (__bf16*)(ws + nxt((size_t)NPTOT * 128 * 2));
    float*  lF  = (float*) (ws + nxt((size_t)NPTOT * 128 * 4));
    __bf16* g1T = (__bf16*)(ws + nxt((size_t)NPTOT * 256 * 2));
    float*  gmx = (float*) (ws + nxt(BATCH * 1024 * 4));

    // 1) weights -> bf16
    k_cvt_bf16<<<(128 * 64 + 255) / 256, 256, 0, stream>>>(W2, wW2, 128 * 64);
    k_cvt_bf16<<<(256 * 128 + 255) / 256, 256, 0, stream>>>(G1, wG1, 256 * 128);
    k_cvt_bf16<<<(1024 * 256 + 255) / 256, 256, 0, stream>>>(G2, wG2, 1024 * 256);

    // 2) layer1 (3->64 relu), point-major bf16
    k_l1<<<NPTOT / 256, 256, 0, stream>>>(pc, W1, b1, hT);

    // 3) layer2 (64->128 relu): 8 oc-tiles; 4 waves x MT=2
    k_gemm_bf16<2><<<dim3(NPTOT / 16, 1), 128, 0, stream>>>(
        wW2, b2, hT, 64, 128, lT, lF, nullptr, 1, NPTS);

    // 4) global layer1 (128->256 relu): 16 oc-tiles; 4 waves x MT=4
    k_gemm_bf16<4><<<dim3(NPTOT / 16, 1), 128, 0, stream>>>(
        wG1, g1, lT, 128, 256, g1T, nullptr, nullptr, 1, NPTS);

    // 5) global layer2 (256->1024 linear) fused with max-over-N reduction
    k_init_gmax<<<(BATCH * 1024 + 255) / 256, 256, 0, stream>>>((unsigned*)gmx, BATCH * 1024);
    k_gemm_bf16<4><<<dim3(NPTOT / 16, 4), 128, 0, stream>>>(
        wG2, g2, g1T, 256, 1024, nullptr, nullptr, gmx, 0, NPTS);

    // 6) out[:,0:1024,:] = gmax broadcast (gather of a broadcast == broadcast)
    {
        long long total = (long long)BATCH * 1024 * (NPTS / 4);
        k_broadcast<<<(unsigned)((total + 255) / 256), 256, 0, stream>>>(gmx, out);
    }

    // 7) KNN top-3 + gather/mean of 128 local channels -> out[:,1024:1152,:]
    k_knn_gather<<<dim3(NPTS / 256, BATCH), 256, 0, stream>>>(pc, qp, lF, out);
}
